// GATConv_multi_66907000537769
// MI455X (gfx1250) — compile-verified, matrix operand order checked by reading, not run on hardware
//
#include <hip/hip_runtime.h>
#include <hip/hip_bf16.h>

#define HEAD    4
#define OUT_DIM 32
#define IN_DIM  256
#define FC_OUT  128            // HEAD * OUT_DIM
#define ALPHA   0.2f

typedef __attribute__((ext_vector_type(16))) _Float16 v16h;
typedef __attribute__((ext_vector_type(8)))  _Float16 v8h;
typedef __attribute__((ext_vector_type(8)))  float    v8f;

// ---------------------------------------------------------------------------
// Kernel 1: Xp = X @ W^T   (N x 256) * (256 x 128) -> (N x 128), fp32 out.
// W staged once per workgroup into LDS as f16 (64 KB of 320 KB WGP LDS).
// Each wave computes 16 rows x 128 cols: 8 f32 accumulators, 64 WMMAs.
// A-stream is software-pipelined (register double buffer) so the 4
// global_load_b128 per K-step overlap the previous step's WMMAs.
// ---------------------------------------------------------------------------
__global__ __launch_bounds__(256)
void gat_gemm_kernel(const float* __restrict__ X, const float* __restrict__ W,
                     float* __restrict__ Xp, int N)
{
    __shared__ _Float16 Wl[FC_OUT * IN_DIM];   // 64 KB, [n][k] row-major f16

    const int tid = threadIdx.x;

    const float4* W4 = (const float4*)W;                 // 8192 float4
    for (int i = tid; i < (FC_OUT * IN_DIM) / 4; i += 256) {
        float4 w = W4[i];
        _Float16* p = &Wl[i * 4];
        p[0] = (_Float16)w.x; p[1] = (_Float16)w.y;
        p[2] = (_Float16)w.z; p[3] = (_Float16)w.w;
    }
    __syncthreads();

    const int lane = tid & 31;
    const int wv   = tid >> 5;
    const int l16  = lane & 15;
    const int half = lane >> 4;
    const int kb   = half * 8;                // K-base per 16-bit A/B layout
    const int mo   = half * 8;                // C/D row offset per layout

    const int m0 = blockIdx.x * 128 + wv * 16;

    int mA = m0 + l16;
    if (mA > N - 1) mA = N - 1;               // branchless clamp: EXEC stays all-1s
    const float* __restrict__ xrow = X + (size_t)mA * IN_DIM;

    v8f acc[8];
#pragma unroll
    for (int t = 0; t < 8; ++t) acc[t] = (v8f){0,0,0,0,0,0,0,0};

    // pipeline prologue: K-step 0 in flight
    float4 cx0 = *(const float4*)(xrow + kb);
    float4 cx1 = *(const float4*)(xrow + kb + 4);
    float4 cx2 = *(const float4*)(xrow + kb + 16);
    float4 cx3 = *(const float4*)(xrow + kb + 20);

#pragma unroll
    for (int kk = 0; kk < IN_DIM; kk += 32) {
        float4 nx0, nx1, nx2, nx3;
        if (kk + 32 < IN_DIM) {
            const float* nxt = xrow + kk + 32 + kb;
            __builtin_prefetch(nxt + 32, 0, 1);          // global_prefetch_b8
            nx0 = *(const float4*)(nxt);
            nx1 = *(const float4*)(nxt + 4);
            nx2 = *(const float4*)(nxt + 16);
            nx3 = *(const float4*)(nxt + 20);
        }

        v16h a;
        a[0]  = (_Float16)cx0.x; a[1]  = (_Float16)cx0.y;
        a[2]  = (_Float16)cx0.z; a[3]  = (_Float16)cx0.w;
        a[4]  = (_Float16)cx1.x; a[5]  = (_Float16)cx1.y;
        a[6]  = (_Float16)cx1.z; a[7]  = (_Float16)cx1.w;
        a[8]  = (_Float16)cx2.x; a[9]  = (_Float16)cx2.y;
        a[10] = (_Float16)cx2.z; a[11] = (_Float16)cx2.w;
        a[12] = (_Float16)cx3.x; a[13] = (_Float16)cx3.y;
        a[14] = (_Float16)cx3.z; a[15] = (_Float16)cx3.w;

#pragma unroll
        for (int t = 0; t < 8; ++t) {
            const _Float16* wr = &Wl[(16 * t + l16) * IN_DIM + kk + kb];
            v8h lo = *(const v8h*)(wr);
            v8h hi = *(const v8h*)(wr + 16);
            v16h b = __builtin_shufflevector(lo, hi,
                       0,1,2,3,4,5,6,7,8,9,10,11,12,13,14,15);
            acc[t] = __builtin_amdgcn_wmma_f32_16x16x32_f16(
                        false, a, false, b, (short)0, acc[t], false, false);
        }

        cx0 = nx0; cx1 = nx1; cx2 = nx2; cx3 = nx3;
    }

#pragma unroll
    for (int t = 0; t < 8; ++t) {
#pragma unroll
        for (int j = 0; j < 8; ++j) {
            int r = m0 + mo + j;
            if (r < N)
                Xp[(size_t)r * FC_OUT + 16 * t + l16] = acc[t][j];
        }
    }
}

// ---------------------------------------------------------------------------
// Kernel 2: per-node attention half-scores  s0[h][n], s1[h][n]
// ---------------------------------------------------------------------------
__global__ __launch_bounds__(256)
void gat_scores_kernel(const float* __restrict__ Xp,
                       const float* __restrict__ a0, const float* __restrict__ a1,
                       float* __restrict__ s0, float* __restrict__ s1, int N)
{
    int tid = blockIdx.x * blockDim.x + threadIdx.x;
    int h = tid & 3;
    int n = tid >> 2;
    if (n >= N) return;

    const float4* xr  = (const float4*)(Xp + (size_t)n * FC_OUT + h * OUT_DIM);
    const float4* a0r = (const float4*)(a0 + h * OUT_DIM);
    const float4* a1r = (const float4*)(a1 + h * OUT_DIM);
    float v0 = 0.f, v1 = 0.f;
#pragma unroll
    for (int i = 0; i < OUT_DIM / 4; ++i) {
        float4 x = xr[i], p = a0r[i], q = a1r[i];
        v0 += x.x * p.x + x.y * p.y + x.z * p.z + x.w * p.w;
        v1 += x.x * q.x + x.y * q.y + x.z * q.z + x.w * q.w;
    }
    s0[h * N + n] = v0;
    s1[h * N + n] = v1;
}

// ---------------------------------------------------------------------------
// Kernel 3: edge aggregation with async global->LDS gather staging.
// One wave per contiguous (sorted) edge range.  Each edge needs the 128-float
// Xp[col] row: one global_load_async_to_lds_b128 (32 lanes x 16B = 512B)
// stages it into a per-wave LDS buffer, double-buffered one edge ahead and
// synchronized with s_wait_asynccnt (async loads complete in order).
// Register accumulation per destination row; atomic flush on row change.
// ---------------------------------------------------------------------------
__global__ __launch_bounds__(256)
void gat_edge_kernel(const float* __restrict__ Xp,
                     const float* __restrict__ s0, const float* __restrict__ s1,
                     const int* __restrict__ row, const int* __restrict__ col,
                     float* __restrict__ out, float* __restrict__ rsum,
                     int N, int E, int eper)
{
    __shared__ float xbuf[8][2][FC_OUT];     // 8 KB: 8 waves x 2 buffers x 512B

    const int lane = threadIdx.x & 31;
    const int wv   = threadIdx.x >> 5;
    const int wid  = blockIdx.x * 8 + wv;
    int e0 = wid * eper;
    if (e0 >= E) return;
    int e1 = e0 + eper; if (e1 > E) e1 = E;

    float* b0 = &xbuf[wv][0][0];
    float* b1 = &xbuf[wv][1][0];
    // Per-lane LDS byte addresses (flat-address low 32 bits == LDS offset).
    unsigned lds0 = (unsigned)(uintptr_t)b0 + (unsigned)lane * 16u;
    unsigned lds1 = (unsigned)(uintptr_t)b1 + (unsigned)lane * 16u;

    const size_t hs = (size_t)N * OUT_DIM;

    // prologue: stage Xp[col[e0]] into buffer 0
    {
        unsigned long long ga = (unsigned long long)(uintptr_t)
            (Xp + (size_t)col[e0] * FC_OUT) + (unsigned long long)lane * 16u;
        asm volatile("global_load_async_to_lds_b128 %0, %1, off"
                     :: "v"(lds0), "v"(ga) : "memory");
    }

    float acc0 = 0.f, acc1 = 0.f, acc2 = 0.f, acc3 = 0.f;
    float as0 = 0.f, as1 = 0.f, as2 = 0.f, as3 = 0.f;
    int cur = row[e0];

    for (int e = e0; e < e1; ++e) {
        int r = row[e];
        int c = col[e];
        int p = (e - e0) & 1;

        // stage next edge's Xp row into the other buffer (redundant re-issue
        // on the last iteration keeps the wait count uniform)
        int ne = (e + 1 < e1) ? e + 1 : e;
        int nc = col[ne];
        unsigned nlds = p ? lds0 : lds1;
        unsigned long long ga = (unsigned long long)(uintptr_t)
            (Xp + (size_t)nc * FC_OUT) + (unsigned long long)lane * 16u;
        asm volatile("global_load_async_to_lds_b128 %0, %1, off"
                     :: "v"(nlds), "v"(ga) : "memory");
        // <=1 outstanding => the gather for edge e has landed in LDS
        asm volatile("s_wait_asynccnt 0x1" ::: "memory");

        if (r != cur) {
            float* o = out + (size_t)cur * OUT_DIM + lane;
            atomicAdd(o,          acc0);
            atomicAdd(o + hs,     acc1);
            atomicAdd(o + 2 * hs, acc2);
            atomicAdd(o + 3 * hs, acc3);
            float sv = (lane == 0) ? as0 : (lane == 1) ? as1
                     : (lane == 2) ? as2 : as3;
            if (lane < 4) atomicAdd(rsum + lane * N + cur, sv);
            acc0 = acc1 = acc2 = acc3 = 0.f;
            as0 = as1 = as2 = as3 = 0.f;
            cur = r;
        }

        // attention weight per head on lanes 0..3, broadcast to the wave
        float att = 0.f;
        if (lane < 4) {
            float sv = s0[lane * N + r] + s1[lane * N + c];
            sv = sv > 0.f ? sv : ALPHA * sv;          // LeakyReLU
            att = __expf(sv);
        }
        float a0  = __shfl(att, 0, 32);
        float a1v = __shfl(att, 1, 32);
        float a2  = __shfl(att, 2, 32);
        float a3  = __shfl(att, 3, 32);

        const float* xc = p ? b1 : b0;                // LDS, conflict-free
        acc0 += a0  * xc[lane];
        acc1 += a1v * xc[32 + lane];
        acc2 += a2  * xc[64 + lane];
        acc3 += a3  * xc[96 + lane];
        as0 += a0; as1 += a1v; as2 += a2; as3 += a3;
    }

    float* o = out + (size_t)cur * OUT_DIM + lane;
    atomicAdd(o,          acc0);
    atomicAdd(o + hs,     acc1);
    atomicAdd(o + 2 * hs, acc2);
    atomicAdd(o + 3 * hs, acc3);
    float sv = (lane == 0) ? as0 : (lane == 1) ? as1
             : (lane == 2) ? as2 : as3;
    if (lane < 4) atomicAdd(rsum + lane * N + cur, sv);
}

// ---------------------------------------------------------------------------
// Kernel 4: softmax normalization   out[h,n,d] /= max(rsum[h,n], 1e-12)
// ---------------------------------------------------------------------------
__global__ __launch_bounds__(256)
void gat_norm_kernel(float* __restrict__ out, const float* __restrict__ rsum, int N)
{
    int tid = blockIdx.x * blockDim.x + threadIdx.x;
    int total = HEAD * N * OUT_DIM;
    if (tid >= total) return;
    int hn = tid >> 5;                       // tid = (h*N + n)*32 + d
    float den = fmaxf(rsum[hn], 1e-12f);
    out[tid] = out[tid] / den;
}

// ---------------------------------------------------------------------------
extern "C" void kernel_launch(void* const* d_in, const int* in_sizes, int n_in,
                              void* d_out, int out_size, void* d_ws, size_t ws_size,
                              hipStream_t stream)
{
    const float* X   = (const float*)d_in[0];
    const float* W   = (const float*)d_in[1];
    const float* a0  = (const float*)d_in[2];
    const float* a1  = (const float*)d_in[3];
    const int*   row = (const int*)d_in[4];
    const int*   col = (const int*)d_in[5];
    float*       out = (float*)d_out;

    const int N = in_sizes[0] / IN_DIM;      // 50000
    const int E = in_sizes[4];               // 800000

    float* Xp   = (float*)d_ws;                      // N*128
    float* s0   = Xp + (size_t)N * FC_OUT;           // 4*N
    float* s1   = s0 + (size_t)HEAD * N;             // 4*N
    float* rsum = s1 + (size_t)HEAD * N;             // 4*N

    hipMemsetAsync(out,  0, (size_t)out_size * sizeof(float), stream);
    hipMemsetAsync(rsum, 0, (size_t)HEAD * N * sizeof(float), stream);

    int gemm_blocks = (N + 127) / 128;
    gat_gemm_kernel<<<gemm_blocks, 256, 0, stream>>>(X, W, Xp, N);

    int sc_blocks = (HEAD * N + 255) / 256;
    gat_scores_kernel<<<sc_blocks, 256, 0, stream>>>(Xp, a0, a1, s0, s1, N);

    const int total_waves = 4096;
    int eper = (E + total_waves - 1) / total_waves;
    gat_edge_kernel<<<total_waves / 8, 256, 0, stream>>>(Xp, s0, s1, row, col,
                                                         out, rsum, N, E, eper);

    int nm_blocks = (HEAD * N * OUT_DIM + 255) / 256;
    gat_norm_kernel<<<nm_blocks, 256, 0, stream>>>(out, rsum, N);
}
